// AttDP_11879879544256
// MI455X (gfx1250) — matrix operand order repaired and checked by reading, not compiled, not run on hardware
//
#include <hip/hip_runtime.h>
#include <hip/hip_bf16.h>

#define BATCH  32
#define SEQ    1024
#define HIDDEN 512
#define NEGBIG -1000000000.0f

typedef __attribute__((ext_vector_type(16))) __bf16 v16bf;
typedef __attribute__((ext_vector_type(8)))  __bf16 v8bf;
typedef __attribute__((ext_vector_type(8)))  float  v8f;

// -------------------------------------------------------------------------
// Kernel 0: Wt[n][k] = bf16(W[k][n])   (512x512, one-time, L2-resident)
// Classic 32x32 LDS tile transpose, 256 threads/block, grid (16,16).
// -------------------------------------------------------------------------
__global__ __launch_bounds__(256) void transpose_w_bf16(
    const float* __restrict__ W, __bf16* __restrict__ Wt)
{
    __shared__ float t[32][33];                 // +1 pad: no bank conflicts
    const int tx = threadIdx.x & 31;
    const int ty = threadIdx.x >> 5;            // 0..7
    const int k0 = blockIdx.y * 32;
    const int n0 = blockIdx.x * 32;

#pragma unroll
    for (int j = 0; j < 4; ++j)                 // coalesced read of W rows
        t[ty + 8 * j][tx] = W[(size_t)(k0 + ty + 8 * j) * HIDDEN + n0 + tx];
    __syncthreads();
#pragma unroll
    for (int j = 0; j < 4; ++j)                 // coalesced write of Wt rows
        Wt[(size_t)(n0 + ty + 8 * j) * HIDDEN + k0 + tx] =
            (__bf16)t[tx][ty + 8 * j];
}

// -------------------------------------------------------------------------
// Kernel 1: Y = bf16( relu( X @ W + bias ) ) using pre-transposed bf16 Wt.
//   X: (M, 512) f32 row-major, Wt: (512, 512) bf16 (Wt[n][k]), Y: (M,512) bf16
// Block = 256 threads (8 waves). Block tile = 128 rows x 64 cols.
// B fragment = one contiguous 32-byte bf16 load per lane (no LDS, no cvt).
// -------------------------------------------------------------------------
__global__ __launch_bounds__(256) void proj_relu_bf16(
    const float* __restrict__ X, const __bf16* __restrict__ Wt,
    const float* __restrict__ bias, __bf16* __restrict__ Y)
{
    const int lane = threadIdx.x & 31;
    const int wave = threadIdx.x >> 5;
    const int half = lane >> 4;   // 0: lanes 0-15, 1: lanes 16-31
    const int l16  = lane & 15;

    const int m0  = blockIdx.x * 128 + wave * 16;
    const int n0  = blockIdx.y * 64;
    const int row = m0 + l16;

    v8f acc[4];
#pragma unroll
    for (int i = 0; i < 4; ++i) acc[i] = (v8f){0.f,0.f,0.f,0.f,0.f,0.f,0.f,0.f};

    for (int kk = 0; kk < HIDDEN; kk += 32) {
        // ---- A fragment (16x32 bf16): elems 0-15 -> K=half*16+0..15 ----
        const float* ap = X + (size_t)row * HIDDEN + kk + half * 16;
        float4 a0 = ((const float4*)ap)[0];
        float4 a1 = ((const float4*)ap)[1];
        float4 a2 = ((const float4*)(ap + 8))[0];
        float4 a3 = ((const float4*)(ap + 8))[1];
        v16bf a;
        a[0]  = (__bf16)a0.x; a[1]  = (__bf16)a0.y; a[2]  = (__bf16)a0.z; a[3]  = (__bf16)a0.w;
        a[4]  = (__bf16)a1.x; a[5]  = (__bf16)a1.y; a[6]  = (__bf16)a1.z; a[7]  = (__bf16)a1.w;
        a[8]  = (__bf16)a2.x; a[9]  = (__bf16)a2.y; a[10] = (__bf16)a2.z; a[11] = (__bf16)a2.w;
        a[12] = (__bf16)a3.x; a[13] = (__bf16)a3.y; a[14] = (__bf16)a3.z; a[15] = (__bf16)a3.w;

        // ---- B fragments: lane holds col n = n-tile + l16,
        //      elems contiguous in k (kk + half*16 + 0..15) ----
#pragma unroll
        for (int i = 0; i < 4; ++i) {
            int n = n0 + i * 16 + l16;
            v16bf bb = *(const v16bf*)(Wt + (size_t)n * HIDDEN + kk + half * 16);
            acc[i] = __builtin_amdgcn_wmma_f32_16x16x32_bf16(
                false, a, false, bb, (short)0, acc[i], false, false);
        }
    }

    // ---- epilogue: bias + relu + bf16 store ----
#pragma unroll
    for (int i = 0; i < 4; ++i) {
        int   n  = n0 + i * 16 + l16;
        float bn = bias[n];
#pragma unroll
        for (int r = 0; r < 8; ++r) {
            int   m = m0 + r + 8 * half;   // C/D layout: VGPR r, lane-half
            float v = acc[i][r] + bn;
            v = v > 0.f ? v : 0.f;
            Y[(size_t)m * HIDDEN + n] = (__bf16)v;
        }
    }
}

// -------------------------------------------------------------------------
// Kernel 2: logits = Qb @ Kb^T / sqrt(512); masked softmax; * query_mask.
// One block per (batch, 16-query-row tile). 8 waves; wave w computes key
// tiles [8w, 8w+8) -> full 16x1024 logits tile in 64KB LDS, then per-row
// wave32 shuffle-reduced softmax and coalesced f32 writeout.
// -------------------------------------------------------------------------
__global__ __launch_bounds__(256) void attn_softmax(
    const __bf16* __restrict__ Q, const __bf16* __restrict__ K,
    const int* __restrict__ qmask, const int* __restrict__ kmask,
    float* __restrict__ out)
{
    __shared__ float sc[16 * SEQ];   // 64 KB of the 320 KB WGP LDS

    const int lane = threadIdx.x & 31;
    const int wave = threadIdx.x >> 5;
    const int half = lane >> 4;
    const int l16  = lane & 15;

    const int b  = blockIdx.x >> 6;       // 64 q-tiles per batch
    const int qt = blockIdx.x & 63;
    const int q0 = qt * 16;

    const __bf16* qp = Q + ((size_t)b * SEQ + q0) * HIDDEN;
    const __bf16* kp = K + (size_t)b * SEQ * HIDDEN;

    v8f acc[8];
#pragma unroll
    for (int i = 0; i < 8; ++i) acc[i] = (v8f){0.f,0.f,0.f,0.f,0.f,0.f,0.f,0.f};

    for (int kk = 0; kk < HIDDEN; kk += 32) {
        // A fragment from bf16 Q (K=half*16+0..15, two 16B chunks per lane)
        const __bf16* ap = qp + (size_t)l16 * HIDDEN + kk + half * 16;
        v8bf lo = *(const v8bf*)ap;
        v8bf hi = *(const v8bf*)(ap + 8);
        v16bf a;
#pragma unroll
        for (int e = 0; e < 8; ++e) { a[e] = lo[e]; a[8 + e] = hi[e]; }

        // B = K^T slice: column j of K^T is row j of K -> contiguous
        // 32-byte load per lane (lane holds key row (tile*16+l16),
        // elem e -> K-dim index kk + half*16 + e).
#pragma unroll
        for (int i = 0; i < 8; ++i) {
            int krow = (wave * 8 + i) * 16 + l16;
            v16bf bb = *(const v16bf*)(kp + (size_t)krow * HIDDEN + kk + half * 16);
            acc[i] = __builtin_amdgcn_wmma_f32_16x16x32_bf16(
                false, a, false, bb, (short)0, acc[i], false, false);
        }
    }

    // scatter logits tile into LDS
#pragma unroll
    for (int i = 0; i < 8; ++i) {
        int cb = (wave * 8 + i) * 16 + l16;
#pragma unroll
        for (int r = 0; r < 8; ++r)
            sc[(r + 8 * half) * SEQ + cb] = acc[i][r];
    }
    __syncthreads();

    // masked softmax: each wave owns rows {2*wave, 2*wave+1}
    const float scale = 0.044194173824159220f;   // 1/sqrt(512)
    const int*  km    = kmask + (size_t)b * SEQ;

    for (int rr = 0; rr < 2; ++rr) {
        int    r  = wave * 2 + rr;
        float* sr = sc + r * SEQ;

        float mx = -3.4e38f;
        for (int j = 0; j < SEQ / 32; ++j) {
            int   col = lane + 32 * j;
            float s   = sr[col] * scale + (km[col] ? 0.0f : NEGBIG);
            sr[col]   = s;
            mx = fmaxf(mx, s);
        }
#pragma unroll
        for (int off = 16; off; off >>= 1)
            mx = fmaxf(mx, __shfl_xor(mx, off, 32));

        float sum = 0.f;
        for (int j = 0; j < SEQ / 32; ++j) {
            int   col = lane + 32 * j;
            float e   = __expf(sr[col] - mx);
            sr[col]   = e;
            sum += e;
        }
#pragma unroll
        for (int off = 16; off; off >>= 1)
            sum += __shfl_xor(sum, off, 32);

        int    qm  = qmask[(size_t)b * SEQ + q0 + r];
        float  fac = qm ? (1.0f / sum) : 0.0f;
        float* op  = out + ((size_t)b * SEQ + q0 + r) * SEQ;
        for (int j = 0; j < SEQ / 32; ++j) {
            int col = lane + 32 * j;
            op[col] = sr[col] * fac;
        }
    }
}

// -------------------------------------------------------------------------
extern "C" void kernel_launch(void* const* d_in, const int* in_sizes, int n_in,
                              void* d_out, int out_size, void* d_ws, size_t ws_size,
                              hipStream_t stream)
{
    const float* query = (const float*)d_in[0];
    const float* key   = (const float*)d_in[1];
    const int*   qmask = (const int*)  d_in[2];
    const int*   kmask = (const int*)  d_in[3];
    const float* Wq    = (const float*)d_in[4];
    const float* bq    = (const float*)d_in[5];
    const float* Wk    = (const float*)d_in[6];
    const float* bk    = (const float*)d_in[7];
    float*       out   = (float*)d_out;

    // workspace layout:
    //   [0, 32MB)      bf16 projected Q
    //   [32MB, 64MB)   bf16 projected K
    //   [64MB, +512KB) bf16 Wq^T
    //   [.. , +512KB)  bf16 Wk^T
    const size_t projBytes = (size_t)BATCH * SEQ * HIDDEN * sizeof(__bf16);
    const size_t wBytes    = (size_t)HIDDEN * HIDDEN * sizeof(__bf16);
    __bf16* qws = (__bf16*)d_ws;
    __bf16* kws = (__bf16*)((char*)d_ws + projBytes);
    __bf16* wqt = (__bf16*)((char*)d_ws + 2 * projBytes);
    __bf16* wkt = (__bf16*)((char*)d_ws + 2 * projBytes + wBytes);

    dim3 gT(HIDDEN / 32, HIDDEN / 32);            // (16,16)
    transpose_w_bf16<<<gT, 256, 0, stream>>>(Wq, wqt);
    transpose_w_bf16<<<gT, 256, 0, stream>>>(Wk, wkt);

    dim3 gProj(BATCH * SEQ / 128, HIDDEN / 64);   // (256, 8)
    proj_relu_bf16<<<gProj, 256, 0, stream>>>(query, wqt, bq, qws);
    proj_relu_bf16<<<gProj, 256, 0, stream>>>(key,   wkt, bk, kws);

    dim3 gAtt(BATCH * SEQ / 16);                  // 2048 blocks
    attn_softmax<<<gAtt, 256, 0, stream>>>(qws, kws, qmask, kmask, out);
}
